// BitNetAttention_89489938580086
// MI455X (gfx1250) — compile-verified, matrix operand order checked by reading, not run on hardware
//
#include <hip/hip_runtime.h>
#include <hip/hip_bf16.h>
#include <cstdint>

typedef __attribute__((ext_vector_type(16))) _Float16 v16h;
typedef __attribute__((ext_vector_type(8)))  _Float16 v8h;
typedef __attribute__((ext_vector_type(8)))  float    v8f;
typedef int v4i __attribute__((vector_size(4 * sizeof(int))));

#define DEVFN __device__ __forceinline__
#define AS1 __attribute__((address_space(1)))
#define AS3 __attribute__((address_space(3)))

#if __has_builtin(__builtin_amdgcn_global_load_async_to_lds_b128) && \
    __has_builtin(__builtin_amdgcn_s_wait_asynccnt)
#define HAVE_ASYNC 1
#else
#define HAVE_ASYNC 0
#endif

constexpr int BB  = 2;
constexpr int S   = 2048;
constexpr int H   = 1024;
constexpr int NH  = 16;
constexpr int HD  = 64;
constexpr int N   = BB * S;           // 4096 rows total
constexpr float ATTN_SCALE = 0.125f;  // 1/sqrt(64)

// Attention dynamic-LDS layout (bytes)
constexpr int SC_BYTES   = 16 * S * 4;                 // 131072: scores [16][S] f32
constexpr int RED_OFF    = SC_BYTES;                   // [16][8] f32
constexpr int STAT_OFF   = RED_OFF + 16 * 8 * 4;       // [16] f32
constexpr int KBUF_OFF   = 132096;                     // 129 KB (aligned past scratch)
constexpr int KBUF_BYTES = 4 * 2 * 16 * 64 * 2;        // 4 waves x 2 bufs x 16x64 f16 = 16 KB
constexpr int VBUF_OFF   = KBUF_OFF + KBUF_BYTES;
constexpr int VBUF_BYTES = 4 * 2 * 16 * 32 * 2;        // 8 KB
constexpr int SMEM_TOTAL = VBUF_OFF + VBUF_BYTES;      // ~153 KB (< 320 KB WGP LDS)

// ---------------------------------------------------------------------------
// WMMA helpers (V_WMMA_F32_16X16X32_F16, wave32)
// ---------------------------------------------------------------------------
DEVFN v8f wmma_f16(v16h a, v16h b, v8f c) {
  return __builtin_amdgcn_wmma_f32_16x16x32_f16(false, a, false, b,
                                                (short)0, c, false, false);
}

// 16-bit A (16x32, lane = M row) per ISA 7.12.2; B (32x16, lane = N col) is
// the mirrored packing, so a row-major [rows x K] matrix serves both.
DEVFN v16h load_frag16(const _Float16* __restrict__ base, int row, int ld,
                       int k0, int lane) {
  const int r   = row + (lane & 15);
  const int off = (lane >> 4) << 3;
  const _Float16* p = base + (size_t)r * ld + (k0 + off);
  v8h lo = *(const v8h*)(p);        // b128
  v8h hi = *(const v8h*)(p + 16);   // b128
  v16h f;
#pragma unroll
  for (int i = 0; i < 8; ++i) { f[i] = lo[i]; f[i + 8] = hi[i]; }
  return f;
}

// Same fragment sourced from an f32 LDS buffer (converted on the fly).
DEVFN v16h load_frag_f32(const float* __restrict__ s, int ld, int k0, int lane) {
  const int r   = lane & 15;
  const int off = (lane >> 4) << 3;
  const float* p = s + (size_t)r * ld + (k0 + off);
  v16h f;
#pragma unroll
  for (int i = 0; i < 8; ++i) {
    f[i]     = (_Float16)p[i];
    f[i + 8] = (_Float16)p[i + 16];
  }
  return f;
}

// ---------------------------------------------------------------------------
// Async global->LDS staging (ASYNCcnt pipeline); falls back to plain stores.
// ---------------------------------------------------------------------------
DEVFN void async_copy16(const _Float16* g, _Float16* l) {
#if HAVE_ASYNC
  __builtin_amdgcn_global_load_async_to_lds_b128(
      (AS1 v4i*)(uintptr_t)g, (AS3 v4i*)(unsigned)(uintptr_t)l, 0, 0);
#else
  *(v8h*)l = *(const v8h*)g;
#endif
}

template <int Ncnt>
DEVFN void wait_async() {
#if HAVE_ASYNC
  __builtin_amdgcn_s_wait_asynccnt((short)Ncnt);
#endif
}

// Stage a 16x64 f16 tile (2 KB): 128 x 16B chunks, 4 per lane.
DEVFN void stage16x64(const _Float16* g, int ldg, _Float16* l, int lane) {
#pragma unroll
  for (int i = 0; i < 4; ++i) {
    const int c  = i * 32 + lane;
    const int r  = c >> 3;            // 8 chunks per 64-half row
    const int co = (c & 7) * 8;
    async_copy16(g + (size_t)r * ldg + co, l + r * 64 + co);
  }
}

// Stage a 16x32 f16 tile (1 KB): 64 x 16B chunks, 2 per lane.
DEVFN void stage16x32(const _Float16* g, int ldg, _Float16* l, int lane) {
#pragma unroll
  for (int i = 0; i < 2; ++i) {
    const int c  = i * 32 + lane;
    const int r  = c >> 2;            // 4 chunks per 32-half row
    const int co = (c & 3) * 8;
    async_copy16(g + (size_t)r * ldg + co, l + r * 32 + co);
  }
}

// ---------------------------------------------------------------------------
// Small prep kernels
// ---------------------------------------------------------------------------
__global__ void zero_alphas(float* a) {
  if (threadIdx.x < 4) a[threadIdx.x] = 0.0f;
}

__global__ void abs_sum4(const float* __restrict__ W0, const float* __restrict__ W1,
                         const float* __restrict__ W2, const float* __restrict__ W3,
                         float* __restrict__ sums) {
  const float* W = (blockIdx.y == 0) ? W0 : (blockIdx.y == 1) ? W1
                 : (blockIdx.y == 2) ? W2 : W3;
  __shared__ float sh[256];
  float acc = 0.0f;
  for (int i = blockIdx.x * 256 + threadIdx.x; i < H * H; i += gridDim.x * 256)
    acc += fabsf(W[i]);
  sh[threadIdx.x] = acc;
  __syncthreads();
  for (int st = 128; st > 0; st >>= 1) {
    if (threadIdx.x < st) sh[threadIdx.x] += sh[threadIdx.x + st];
    __syncthreads();
  }
  if (threadIdx.x == 0) atomicAdd(&sums[blockIdx.y], sh[0]);
}

__global__ void cvt_f16(const float* __restrict__ src, _Float16* __restrict__ dst, int n) {
  int i = blockIdx.x * 256 + threadIdx.x;
  if (i < n) dst[i] = (_Float16)src[i];
}

__global__ void quantize4(const float* __restrict__ W0, const float* __restrict__ W1,
                          const float* __restrict__ W2, const float* __restrict__ W3,
                          _Float16* __restrict__ Wsgn) {
  const float* W = (blockIdx.y == 0) ? W0 : (blockIdx.y == 1) ? W1
                 : (blockIdx.y == 2) ? W2 : W3;
  _Float16* D = Wsgn + (size_t)blockIdx.y * H * H;
  int i = blockIdx.x * 256 + threadIdx.x;
  if (i < H * H) {
    float w = W[i];
    D[i] = (_Float16)((w > 0.0f) ? 1.0f : ((w < 0.0f) ? -1.0f : 0.0f));
  }
}

// Vt[(b*NH+h)*HD + d][s] = V[b*S+s][h*HD+d]
__global__ void transpose_v(const _Float16* __restrict__ V, _Float16* __restrict__ Vt) {
  int idx = blockIdx.x * 256 + threadIdx.x;
  if (idx >= N * H) return;
  int col  = idx & (H - 1);
  int rowN = idx >> 10;
  int b = rowN >> 11, s = rowN & (S - 1);
  int h = col >> 6,  d = col & (HD - 1);
  Vt[((size_t)((b * NH + h) * HD + d) << 11) + s] = V[idx];
}

// ---------------------------------------------------------------------------
// BitLinear GEMM: Y[n,j] = alpha * sum_k X[n,k] * Wsgn[j,k]
// 32x64 tile per wave: 2 A-frags + 4 B-frags -> 8 WMMA per K chunk
// (0.75 fragment loads per WMMA vs 2.0 for a 16x16 tile).
// ---------------------------------------------------------------------------
template <bool F32OUT>
__global__ __launch_bounds__(128) void bitlinear_gemm(
    const _Float16* __restrict__ X, const _Float16* __restrict__ Wsgn,
    const float* __restrict__ alpha_sum, void* __restrict__ out,
    int Hin, int Hout) {
  const int lane = threadIdx.x & 31;
  const int wave = threadIdx.x >> 5;
  const int row  = blockIdx.y * 32;
  const int col  = blockIdx.x * 256 + wave * 64;

  v8f acc[2][4] = {};
  for (int k0 = 0; k0 < Hin; k0 += 32) {
    if (k0 + 64 < Hin) {
      __builtin_prefetch(X    + (size_t)row * Hin + k0 + 64, 0, 1);
      __builtin_prefetch(Wsgn + (size_t)col * Hin + k0 + 64, 0, 1);
    }
    v16h a0 = load_frag16(X, row,      Hin, k0, lane);
    v16h a1 = load_frag16(X, row + 16, Hin, k0, lane);
#pragma unroll
    for (int j = 0; j < 4; ++j) {
      v16h bb = load_frag16(Wsgn, col + j * 16, Hin, k0, lane);
      acc[0][j] = wmma_f16(a0, bb, acc[0][j]);
      acc[1][j] = wmma_f16(a1, bb, acc[1][j]);
    }
  }
  const float alpha = alpha_sum[0] * (1.0f / ((float)H * (float)H));
#pragma unroll
  for (int i = 0; i < 2; ++i) {
#pragma unroll
    for (int j = 0; j < 4; ++j) {
      const int ccol = col + j * 16 + (lane & 15);
      const int rb   = row + i * 16 + ((lane >> 4) << 3);
      if constexpr (F32OUT) {
        float* O = (float*)out;
#pragma unroll
        for (int v = 0; v < 8; ++v)
          O[(size_t)(rb + v) * Hout + ccol] = acc[i][j][v] * alpha;
      } else {
        _Float16* O = (_Float16*)out;
#pragma unroll
        for (int v = 0; v < 8; ++v)
          O[(size_t)(rb + v) * Hout + ccol] = (_Float16)(acc[i][j][v] * alpha);
      }
    }
  }
}

// ---------------------------------------------------------------------------
// Attention: one block (4 waves) per (b, h, 16-row q tile).
// K and Vt tiles are double-buffered through LDS with async copies.
// ---------------------------------------------------------------------------
__global__ __launch_bounds__(128) void attention_kernel(
    const _Float16* __restrict__ Q, const _Float16* __restrict__ K,
    const _Float16* __restrict__ Vt, float* __restrict__ attn,
    _Float16* __restrict__ Ctx) {
  extern __shared__ char smem_raw[];
  float* sc      = (float*)smem_raw;
  float* red     = (float*)(smem_raw + RED_OFF);
  float* rowstat = (float*)(smem_raw + STAT_OFF);

  const int tid  = threadIdx.x;
  const int lane = tid & 31;
  const int wave = tid >> 5;
  _Float16* kb0 = (_Float16*)(smem_raw + KBUF_OFF) + (size_t)wave * (2 * 16 * 64);
  _Float16* kb1 = kb0 + 16 * 64;
  _Float16* vb0 = (_Float16*)(smem_raw + VBUF_OFF) + (size_t)wave * (2 * 16 * 32);
  _Float16* vb1 = vb0 + 16 * 32;

  const int qt = blockIdx.x, h = blockIdx.y, b = blockIdx.z;
  const int qrow = b * S + qt * 16;
  const int hcol = h * HD;

  // ---- scores = (Q K^T) * SCALE ----
  const v16h qa0 = load_frag16(Q, qrow, H, hcol,      lane);
  const v16h qa1 = load_frag16(Q, qrow, H, hcol + 32, lane);
  const _Float16* Kbase = K + (size_t)(b * S) * H + hcol;

  stage16x64(Kbase + (size_t)(wave * 16) * H, H, kb0, lane);
  for (int kt = wave; kt < S / 16; kt += 4) {
    const bool more = (kt + 4) < S / 16;
    if (more) {
      stage16x64(Kbase + (size_t)((kt + 4) * 16) * H, H, kb1, lane);
      wait_async<4>();   // only the just-issued tile may remain in flight
    } else {
      wait_async<0>();
    }
    v8f c = {};
    c = wmma_f16(qa0, load_frag16(kb0, 0, 64, 0,  lane), c);
    c = wmma_f16(qa1, load_frag16(kb0, 0, 64, 32, lane), c);

    const int cc = kt * 16 + (lane & 15);
    const int r0 = (lane >> 4) << 3;
#pragma unroll
    for (int v = 0; v < 8; ++v)
      sc[(size_t)(r0 + v) * S + cc] = c[v] * ATTN_SCALE;

    _Float16* t = kb0; kb0 = kb1; kb1 = t;
  }
  __syncthreads();

  // ---- softmax over 16 rows (8 threads / row, 256 cols each) ----
  const int row = tid >> 3;
  const int seg = tid & 7;
  float* sr = sc + (size_t)row * S + seg * 256;

  float m = -3.402823466e38f;
  for (int j = 0; j < 256; ++j) m = fmaxf(m, sr[j]);
  red[row * 8 + seg] = m;
  __syncthreads();
  if (tid < 16) {
    float mm = red[tid * 8];
    for (int j = 1; j < 8; ++j) mm = fmaxf(mm, red[tid * 8 + j]);
    rowstat[tid] = mm;
  }
  __syncthreads();
  const float mm = rowstat[row];
  float ssum = 0.0f;
  for (int j = 0; j < 256; ++j) {
    float e = __expf(sr[j] - mm);
    sr[j] = e;
    ssum += e;
  }
  red[row * 8 + seg] = ssum;
  __syncthreads();
  if (tid < 16) {
    float s2 = 0.0f;
    for (int j = 0; j < 8; ++j) s2 += red[tid * 8 + j];
    rowstat[tid] = 1.0f / s2;
  }
  __syncthreads();
  const float inv = rowstat[row];
  float* ga = attn + (((size_t)(b * NH + h) * S + (qt * 16 + row)) * S) + seg * 256;
  for (int j = 0; j < 256; ++j) {
    float p = sr[j] * inv;
    sr[j] = p;          // normalized probs stay in LDS for the ctx GEMM
    ga[j] = p;          // attention-matrix output
  }
  __syncthreads();

  // ---- ctx = P . V : wave w owns d-tile [w*16, w*16+16) ----
  v8f c2 = {};
  const int vrow = (b * NH + h) * HD + wave * 16;
  const _Float16* Vbase = Vt + (size_t)vrow * S;

  stage16x32(Vbase, S, vb0, lane);
  for (int kc = 0; kc < S; kc += 32) {
    const bool more = (kc + 32) < S;
    if (more) {
      stage16x32(Vbase + kc + 32, S, vb1, lane);
      wait_async<2>();
    } else {
      wait_async<0>();
    }
    v16h a  = load_frag_f32(sc, S, kc, lane);
    c2 = wmma_f16(a, load_frag16(vb0, 0, 32, 0, lane), c2);
    _Float16* t = vb0; vb0 = vb1; vb1 = t;
  }
  const int ccol = hcol + wave * 16 + (lane & 15);
  const int rb   = qrow + ((lane >> 4) << 3);
#pragma unroll
  for (int v = 0; v < 8; ++v)
    Ctx[(size_t)(rb + v) * H + ccol] = (_Float16)c2[v];
}

// ---------------------------------------------------------------------------
// Host-side orchestration
// ---------------------------------------------------------------------------
extern "C" void kernel_launch(void* const* d_in, const int* in_sizes, int n_in,
                              void* d_out, int out_size, void* d_ws, size_t ws_size,
                              hipStream_t stream) {
  const float* hs = (const float*)d_in[0];
  const float* Wq = (const float*)d_in[1];
  const float* Wk = (const float*)d_in[2];
  const float* Wv = (const float*)d_in[3];
  const float* Wo = (const float*)d_in[4];

  float* out_f32  = (float*)d_out;                 // [B,S,H]
  float* attn_out = out_f32 + (size_t)N * H;       // [B,NH,S,S]

  char* w = (char*)d_ws;
  float* alphas = (float*)w;
  size_t o = 256;
  _Float16* X16  = (_Float16*)(w + o); o += (size_t)N * H * 2;
  _Float16* Wsgn = (_Float16*)(w + o); o += (size_t)4 * H * H * 2;
  _Float16* Q16  = (_Float16*)(w + o); o += (size_t)N * H * 2;
  _Float16* K16  = (_Float16*)(w + o); o += (size_t)N * H * 2;
  _Float16* V16  = (_Float16*)(w + o); o += (size_t)N * H * 2;
  _Float16* Vt16 = (_Float16*)(w + o); o += (size_t)N * H * 2;
  _Float16* C16  = (_Float16*)(w + o); o += (size_t)N * H * 2;

  zero_alphas<<<1, 32, 0, stream>>>(alphas);
  abs_sum4<<<dim3(256, 4), 256, 0, stream>>>(Wq, Wk, Wv, Wo, alphas);

  quantize4<<<dim3((H * H) / 256, 4), 256, 0, stream>>>(Wq, Wk, Wv, Wo, Wsgn);
  cvt_f16<<<(N * H) / 256, 256, 0, stream>>>(hs, X16, N * H);

  bitlinear_gemm<false><<<dim3(H / 256, N / 32), 128, 0, stream>>>(
      X16, Wsgn + 0 * (size_t)H * H, alphas + 0, Q16, H, H);
  bitlinear_gemm<false><<<dim3(H / 256, N / 32), 128, 0, stream>>>(
      X16, Wsgn + 1 * (size_t)H * H, alphas + 1, K16, H, H);
  bitlinear_gemm<false><<<dim3(H / 256, N / 32), 128, 0, stream>>>(
      X16, Wsgn + 2 * (size_t)H * H, alphas + 2, V16, H, H);

  transpose_v<<<(N * H) / 256, 256, 0, stream>>>(V16, Vt16);

  attention_kernel<<<dim3(S / 16, NH, BB), 128, SMEM_TOTAL, stream>>>(
      Q16, K16, Vt16, attn_out, C16);

  bitlinear_gemm<true><<<dim3(H / 256, N / 32), 128, 0, stream>>>(
      C16, Wsgn + 3 * (size_t)H * H, alphas + 3, out_f32, H, H);
}
